// CausalConv1d_40063454937734
// MI455X (gfx1250) — matrix operand order repaired and checked by reading, not compiled, not run on hardware
//
#include <hip/hip_runtime.h>
#include <stdint.h>

// ---------------- types ----------------
typedef __attribute__((ext_vector_type(16))) __bf16 v16bf;
typedef __attribute__((ext_vector_type(8)))  __bf16 v8bf;
typedef __attribute__((ext_vector_type(4)))  __bf16 v4bf;
typedef __attribute__((ext_vector_type(8)))  float  v8f;
typedef __attribute__((ext_vector_type(4)))  unsigned int v4u;
typedef __attribute__((ext_vector_type(4)))  int    v4i;
typedef __attribute__((ext_vector_type(8)))  int    v8i;

// ---------------- problem constants ----------------
#define B_   4
#define L_   4096
#define D_   2048
#define K_   4
#define G_   8
#define CPG  256            // channels per group (D/G)

// ---------------- tiling ----------------
#define TM      128                 // time rows per workgroup
#define TN      256                 // out channels per workgroup (= one full group)
#define XROWS   (TM + K_ - 1)       // 131 x rows (tile + causal halo)
#define XPITCH  (CPG + 8)           // 264 bf16 -> 528B pitch (16B aligned, bank-rotated)
#define WCOLS   64                  // i-columns of w slab resident in LDS
#define WPITCH  (WCOLS + 8)         // 72 bf16 -> 144B pitch (= TDM pad: 32dw row + 4dw pad)
#define NSLAB   (K_ * (CPG / WCOLS))// 16 weight slabs per workgroup

#define SMEM_X_BYTES ((size_t)XROWS * XPITCH * 2)   // 69168
#define SMEM_W_BYTES ((size_t)TN * WPITCH * 2)      // 36864 per buffer
#define SMEM_BYTES   (SMEM_X_BYTES + 2 * SMEM_W_BYTES) // ~140 KB (<320 KB WGP LDS)

// ---------------- CDNA5 feature detection ----------------
#if __has_builtin(__builtin_amdgcn_tensor_load_to_lds)
#define HAVE_TDM 1
#else
#define HAVE_TDM 0
#endif

// amdgpu-toolchain (therock headers) => 6-arg tensor builtin; ROCm 7.2 => 5-arg
#if __has_include(<hip/amd_detail/amd_gfx1250_TDM.h>)
#define TDM_ARITY6 1
#else
#define TDM_ARITY6 0
#endif

__device__ inline uint32_t lds_offset_of(const void* p) {
    // generic -> LDS addrspace; ptrtoint of the 32-bit LDS pointer = byte offset
    return (uint32_t)(uintptr_t)(__attribute__((address_space(3))) const char*)p;
}

__device__ inline void wait_tensorcnt_le(int n) {
#if __has_builtin(__builtin_amdgcn_s_wait_tensorcnt)
    if (n == 0) __builtin_amdgcn_s_wait_tensorcnt(0);
    else        __builtin_amdgcn_s_wait_tensorcnt(1);
#else
    if (n == 0) asm volatile("s_wait_tensorcnt 0x0" ::: "memory");
    else        asm volatile("s_wait_tensorcnt 0x1" ::: "memory");
#endif
}

#if HAVE_TDM
// TDM: load one [256 o][64 i] bf16 slab (row pitch 256 elems in global) into LDS
// with pad_amount=4dw after every pad_interval=32dw -> LDS pitch 72 bf16 = WPITCH.
__device__ inline void tdm_load_w_slab(const __bf16* gsrc, uint32_t lds_off) {
    const uint64_t ga = (uint64_t)(uintptr_t)gsrc;
    v4u g0;
    g0.x = 1u;                                    // count=1 (valid), no gather
    g0.y = lds_off;                               // LDS byte address of tile start
    g0.z = (uint32_t)ga;                          // global_addr[31:0]
    g0.w = (uint32_t)(ga >> 32) | (2u << 30);     // global_addr[56:32] | type=2
    v8i g1;
    g1[0] = (int)((1u << 16)      // data_size = 2 bytes
                | (1u << 20)      // pad_enable
                | (4u << 22)      // pad_interval: 32 DWORDs (=128B row)
                | (3u << 25));    // pad_amount:   4 DWORDs (=16B)
    g1[1] = (int)(256u << 16);    // tensor_dim0[15:0] -> bits 63:48
    g1[2] = (int)(256u << 16);    // tensor_dim0 hi=0 ; tensor_dim1[15:0] -> bits 95:80
    g1[3] = (int)(64u  << 16);    // tensor_dim1 hi=0 ; tile_dim0=64     -> bits 127:112
    g1[4] = 256;                  // tile_dim1=256 ; tile_dim2=0
    g1[5] = 256;                  // tensor_dim0_stride[31:0] = 256 elems
    g1[6] = 0;                    // stride hi ; tensor_dim1_stride lo
    g1[7] = 0;
    v4i z4 = {0, 0, 0, 0};        // groups 2/3 unused (2D tensor)
#if TDM_ARITY6
    v8i z8 = {0, 0, 0, 0, 0, 0, 0, 0};
    __builtin_amdgcn_tensor_load_to_lds(g0, g1, z4, z4, z8, 0);
#else
    __builtin_amdgcn_tensor_load_to_lds(g0, g1, z4, z4, 0);
#endif
}
#endif

// =====================================================================
// Prepass: pack w [K][I][O] fp32  ->  wT [g][k][o_in_group][i] bf16
// =====================================================================
__global__ __launch_bounds__(256) void wt_pack_kernel(const float* __restrict__ w,
                                                      __bf16* __restrict__ wT) {
    int idx = blockIdx.x * 256 + threadIdx.x;          // coalesced over O
    if (idx >= K_ * CPG * D_) return;
    int o = idx % D_;
    int i = (idx / D_) % CPG;
    int k = idx / (D_ * CPG);
    int g  = o >> 8;
    int og = o & (CPG - 1);
    wT[(((size_t)(g * K_ + k) * CPG + og) * CPG) + i] = (__bf16)w[idx];
}

// =====================================================================
// Main kernel: causal grouped conv as tiled bf16 WMMA GEMM
//   out[b, t0+m, g*256+n] = sum_k sum_i x[b, t0+m+k-3, g*256+i] * w[k, i, g*256+n]
// =====================================================================
__global__ __launch_bounds__(256) void cconv_wmma_kernel(const float*  __restrict__ x,
                                                         const __bf16* __restrict__ wT,
                                                         float*        __restrict__ out) {
    extern __shared__ __align__(16) char smem[];
    __bf16* ldsX  = (__bf16*)smem;                                        // [XROWS][XPITCH]
    __bf16* ldsW0 = (__bf16*)(smem + SMEM_X_BYTES);                       // [TN][WPITCH]
    __bf16* ldsW1 = (__bf16*)(smem + SMEM_X_BYTES + SMEM_W_BYTES);       // [TN][WPITCH]

    const int tid   = threadIdx.x;
    const int lane  = tid & 31;
    const int wave  = tid >> 5;          // 8 waves
    const int waveM = wave & 1;          // 2 x 64-row wave tiles
    const int waveN = wave >> 1;         // 4 x 64-col wave tiles

    const int wg    = blockIdx.x;        // 1024 = B * (L/TM) * G
    const int g     = wg & (G_ - 1);
    const int tTile = (wg >> 3) & 31;
    const int b     = wg >> 8;
    const int t0    = tTile * TM;

    const __bf16* wTg = wT + (size_t)(g * K_) * CPG * CPG;

#if HAVE_TDM
    // Kick off slab 0 via the Tensor Data Mover while we stage x.
    if (wave == 0)
        tdm_load_w_slab(wTg, lds_offset_of(ldsW0));
#endif

    // ---------------- stage x tile: fp32 global -> bf16 LDS ----------------
    {
        const float* xg = x + ((size_t)b * L_) * D_ + g * CPG;
        for (int u = tid; u < XROWS * (CPG / 4); u += 256) {
            const int r  = u >> 6;                 // row within tile
            const int c4 = (u & 63) << 2;          // 4-float column
            const int t  = t0 - (K_ - 1) + r;      // causal: row r -> time t0-3+r
            float4 v = make_float4(0.f, 0.f, 0.f, 0.f);
            if (t >= 0)
                v = *(const float4*)(xg + (size_t)t * D_ + c4);
            v4bf pv = { (__bf16)v.x, (__bf16)v.y, (__bf16)v.z, (__bf16)v.w };
            *(v4bf*)(ldsX + r * XPITCH + c4) = pv;     // ds_store_b64
        }
    }

    v8f c[4][4] = {};   // 64x64 f32 accumulators per wave (16 WMMA tiles)

    const int mRow0 = waveM * 64 + (lane & 15);   // A row  (M = lane%16)
    const int nRow0 = waveN * 64 + (lane & 15);   // B row  (N = lane%16)
    const int aSel  = (lane >> 4) << 3;           // A half-wave K offset {0,8}
    const int bSel  = (lane >> 4) << 3;           // B half-wave K offset {0,8}

    for (int s = 0; s < NSLAB; ++s) {
        const int k   = s >> 2;
        const int kkO = (s & 3) * WCOLS;
#if HAVE_TDM
        __bf16* cur = (s & 1) ? ldsW1 : ldsW0;
        if (wave == 0) {
            if (s + 1 < NSLAB) {
                const int k2   = (s + 1) >> 2;
                const int kkO2 = ((s + 1) & 3) * WCOLS;
                // prefetch slab s+1 into the other buffer (overlaps compute of s)
                tdm_load_w_slab(wTg + (size_t)k2 * CPG * CPG + kkO2,
                                lds_offset_of((s & 1) ? ldsW0 : ldsW1));
                wait_tensorcnt_le(1);   // slab s complete (TDM in-order per wave)
            } else {
                wait_tensorcnt_le(0);
            }
        }
        __syncthreads();                // also covers ldsX staging on s==0
#else
        __bf16* cur = ldsW0;
        __syncthreads();                // previous slab fully consumed
        {   // synchronous fallback copy: [256 o][64 i] -> LDS
            const __bf16* src = wTg + (size_t)k * CPG * CPG + (size_t)tid * CPG + kkO;
            __bf16*       dst = cur + tid * WPITCH;
#pragma unroll
            for (int j = 0; j < 8; ++j)
                *(v8bf*)(dst + j * 8) = *(const v8bf*)(src + j * 8);
            __builtin_prefetch(src + WCOLS, 0, 1);   // global_prefetch_b8
        }
        __syncthreads();
#endif

#pragma unroll
        for (int kk2 = 0; kk2 < WCOLS; kk2 += 32) {
            // ---- A fragments: 16x32 bf16, rows shifted by tap k ----
            v16bf a[4];
#pragma unroll
            for (int mi = 0; mi < 4; ++mi) {
                const __bf16* ap =
                    ldsX + (size_t)(mRow0 + mi * 16 + k) * XPITCH + kkO + kk2 + aSel;
                v8bf alo = *(const v8bf*)(ap);        // K = sel+0..7
                v8bf ahi = *(const v8bf*)(ap + 16);   // K = 16+sel+0..7
                a[mi] = __builtin_shufflevector(alo, ahi,
                          0, 1, 2, 3, 4, 5, 6, 7, 8, 9, 10, 11, 12, 13, 14, 15);
            }
            // ---- B fragments: 32x16 bf16 from LDS slab ----
            v16bf bb[4];
#pragma unroll
            for (int ni = 0; ni < 4; ++ni) {
                const __bf16* bp =
                    cur + (size_t)(nRow0 + ni * 16) * WPITCH + kk2 + bSel;
                v8bf blo = *(const v8bf*)(bp);        // K = sel+0..7
                v8bf bhi = *(const v8bf*)(bp + 16);   // K = 16+sel+0..7
                bb[ni] = __builtin_shufflevector(blo, bhi,
                          0, 1, 2, 3, 4, 5, 6, 7, 8, 9, 10, 11, 12, 13, 14, 15);
            }
            // ---- 16 WMMAs per k-step ----
#pragma unroll
            for (int mi = 0; mi < 4; ++mi)
#pragma unroll
                for (int ni = 0; ni < 4; ++ni)
                    c[mi][ni] = __builtin_amdgcn_wmma_f32_16x16x32_bf16(
                        false, a[mi], false, bb[ni],
                        (short)0, c[mi][ni], false, false);
        }

#if HAVE_TDM
        __syncthreads();   // all reads of buf[s&1] done before TDM(s+2) rewrites it
#endif
    }

    // ---------------- writeback: C layout VGPR r -> M = r + 8*(lane>=16) ----------------
    float* og = out + ((size_t)b * L_ + t0) * D_ + g * CPG;
    const int mHi = (lane >> 4) << 3;
#pragma unroll
    for (int mi = 0; mi < 4; ++mi) {
        const int mBase = waveM * 64 + mi * 16 + mHi;
#pragma unroll
        for (int ni = 0; ni < 4; ++ni) {
            const int o = waveN * 64 + ni * 16 + (lane & 15);
#pragma unroll
            for (int r = 0; r < 8; ++r)
                og[(size_t)(mBase + r) * D_ + o] = c[mi][ni][r];
        }
    }
}

// =====================================================================
// launcher
// =====================================================================
extern "C" void kernel_launch(void* const* d_in, const int* in_sizes, int n_in,
                              void* d_out, int out_size, void* d_ws, size_t ws_size,
                              hipStream_t stream) {
    const float* x = (const float*)d_in[0];   // [B, L, D] fp32
    const float* w = (const float*)d_in[1];   // [K, D/G, D] fp32
    float* out = (float*)d_out;               // [B, L, D] fp32
    __bf16* wT = (__bf16*)d_ws;               // packed bf16 weights (4 MB)

    const int nW = K_ * CPG * D_;
    wt_pack_kernel<<<(nW + 255) / 256, 256, 0, stream>>>(w, wT);

    const int nBlocks = B_ * (L_ / TM) * G_;  // 1024
    cconv_wmma_kernel<<<nBlocks, 256, SMEM_BYTES, stream>>>(x, wT, out);
}
